// LinearCrossAttention_7730941133046
// MI455X (gfx1250) — matrix operand order repaired and checked by reading, not compiled, not run on hardware
//
#include <hip/hip_runtime.h>
#include <cstdint>
#include <cstddef>

// ---------- types ----------
typedef __attribute__((ext_vector_type(16))) __bf16 bf16x16;
typedef __attribute__((ext_vector_type(8)))  __bf16 bf16x8;
typedef __attribute__((ext_vector_type(8)))  float  f32x8;
typedef __attribute__((ext_vector_type(8)))  unsigned short u16x8;

// ---------- problem constants ----------
#define CB    4
#define CT    4096
#define CS    4096
#define CD    2048
#define CH    32
#define CHKV  8
#define CDH   64
#define CDKV  512
#define CEPS  1e-5f

// ---------- bf16 helpers (bit-exact RNE) ----------
__device__ __forceinline__ unsigned short f2bf(float f) {
  union { float f; unsigned u; } v; v.f = f;
  unsigned u = v.u;
  unsigned r = u + 0x7FFFu + ((u >> 16) & 1u);
  return (unsigned short)(r >> 16);
}
__device__ __forceinline__ float bf2f(unsigned short h) {
  union { unsigned u; float f; } v; v.u = ((unsigned)h) << 16;
  return v.f;
}

// ---------- CDNA5 async global->LDS (ASYNCcnt path) ----------
// GVS mode: mem = SADDR(64) + VADDR(32) ; LDS dest = VDST vgpr (raw LDS addr)
__device__ __forceinline__ void async_load_b128(unsigned lds_addr, unsigned goff,
                                                const void* base) {
  asm volatile("global_load_async_to_lds_b128 %0, %1, %2"
               :
               : "v"(lds_addr), "v"(goff), "s"(base)
               : "memory");
}
__device__ __forceinline__ void wait_async0() {
  asm volatile("s_wait_asynccnt 0" ::: "memory");
}
// truncate flat (LDS-aperture) pointer -> raw LDS byte address (addr[31:0])
__device__ __forceinline__ unsigned lds_addr_of(const void* p) {
  return (unsigned)(uintptr_t)p;
}

// =====================================================================
// Conversion kernels
// =====================================================================
__global__ __launch_bounds__(256) void cvt_bf16_v4(
    const float* __restrict__ in, unsigned short* __restrict__ out, size_t n4) {
  size_t i = (size_t)blockIdx.x * blockDim.x + threadIdx.x;
  if (i >= n4) return;
  float4 f = reinterpret_cast<const float4*>(in)[i];
  ushort4 o;
  o.x = f2bf(f.x); o.y = f2bf(f.y); o.z = f2bf(f.z); o.w = f2bf(f.w);
  reinterpret_cast<ushort4*>(out)[i] = o;
}

// out[n*K + k] = bf16(in[k*N + n])  (weights: [K,N] -> [N,K])
__global__ __launch_bounds__(256) void cvt_transpose_bf16(
    const float* __restrict__ in, unsigned short* __restrict__ out, int K, int N) {
  int i = blockIdx.x * 256 + threadIdx.x;
  if (i >= K * N) return;
  int k = i / N, n = i - k * N;
  out[(size_t)n * K + k] = f2bf(in[i]);
}

// =====================================================================
// Main WMMA GEMM:  C[M,N] = A[M,K] (bf16 rowmajor) * Bt[N,K]^T (bf16)
//   BM=128 BN=128 BK=32; 256 threads = 8 waves, wave tile 32x64
//   Tile staging: async global->LDS (double-buffered, overlapped with WMMA)
// =====================================================================
#define BM 128
#define BN 128
#define BKK 32
#define LPAD 8
#define LSTR (BKK + LPAD)   // 40 halves per LDS row (80B, 16B aligned)

enum { EPI_NONE = 0, EPI_PHI = 1, EPI_PHI_MASK = 2, EPI_MASK = 3 };

template <int EPI, bool TRANS, typename OUTT>
__global__ __launch_bounds__(256) void wmma_gemm(
    const unsigned short* __restrict__ A, const unsigned short* __restrict__ Bt,
    OUTT* __restrict__ C, const int* __restrict__ mask,
    int M, int N, int K, int ldc) {
  __shared__ __align__(16) unsigned short sA[2][BM * LSTR];
  __shared__ __align__(16) unsigned short sB[2][BN * LSTR];

  const int tid  = threadIdx.x;
  const int lane = tid & 31;
  const int wid  = tid >> 5;
  const int waveM = wid & 3;   // 4 wave-rows  -> 32 M each
  const int waveN = wid >> 2;  // 2 wave-cols  -> 64 N each
  const int m0 = blockIdx.y * BM;
  const int n0 = blockIdx.x * BN;

  const unsigned short* Ab = A  + (size_t)m0 * K;
  const unsigned short* Bb = Bt + (size_t)n0 * K;

  // per-thread tile-staging coordinates: 512 16B-chunks per tile, 2 per thread
  unsigned goff[2];
  unsigned ldsAaddr[2][2], ldsBaddr[2][2];
#pragma unroll
  for (int i = 0; i < 2; ++i) {
    const int c = tid + i * 256;
    const int row = c >> 2, c8 = (c & 3) * 8;
    goff[i] = (unsigned)((row * K + c8) * 2);   // bytes from block base
    ldsAaddr[0][i] = lds_addr_of(&sA[0][row * LSTR + c8]);
    ldsAaddr[1][i] = lds_addr_of(&sA[1][row * LSTR + c8]);
    ldsBaddr[0][i] = lds_addr_of(&sB[0][row * LSTR + c8]);
    ldsBaddr[1][i] = lds_addr_of(&sB[1][row * LSTR + c8]);
  }
  auto issue_tile = [&](int buf, int kpos) {
    const unsigned kb = (unsigned)(kpos * 2);
#pragma unroll
    for (int i = 0; i < 2; ++i) {
      async_load_b128(ldsAaddr[buf][i], goff[i] + kb, Ab);
      async_load_b128(ldsBaddr[buf][i], goff[i] + kb, Bb);
    }
  };

  issue_tile(0, 0);
  wait_async0();
  __syncthreads();

  f32x8 acc[2][4] = {};
  const int nk = K / BKK;

  for (int kk = 0; kk < nk; ++kk) {
    const int cur = kk & 1, nxt = cur ^ 1;
    if (kk + 1 < nk) issue_tile(nxt, (kk + 1) * BKK);   // async fill next buffer

    const unsigned short* sa = sA[cur] + (waveM * 32) * LSTR;
    const unsigned short* sb = sB[cur] + (waveN * 64) * LSTR;

    bf16x16 af[2];
#pragma unroll
    for (int i = 0; i < 2; ++i) {
      const unsigned short* p = sa + (i * 16 + (lane & 15)) * LSTR + (lane >> 4) * 8;
      bf16x8 lo = *reinterpret_cast<const bf16x8*>(p);
      bf16x8 hi = *reinterpret_cast<const bf16x8*>(p + 16);
#pragma unroll
      for (int j = 0; j < 8; ++j) { af[i][j] = lo[j]; af[i][j + 8] = hi[j]; }
    }
    bf16x16 bfr[4];
#pragma unroll
    for (int i = 0; i < 4; ++i) {
      const unsigned short* p = sb + (i * 16 + (lane & 15)) * LSTR + (lane >> 4) * 8;
      bf16x8 lo = *reinterpret_cast<const bf16x8*>(p);
      bf16x8 hi = *reinterpret_cast<const bf16x8*>(p + 16);
#pragma unroll
      for (int j = 0; j < 8; ++j) { bfr[i][j] = lo[j]; bfr[i][j + 8] = hi[j]; }
    }
#pragma unroll
    for (int i = 0; i < 2; ++i)
#pragma unroll
      for (int j = 0; j < 4; ++j)
        acc[i][j] = __builtin_amdgcn_wmma_f32_16x16x32_bf16(
            false, af[i], false, bfr[j], (short)0, acc[i][j], false, false);

    if (kk + 1 < nk) wait_async0();   // next-buffer fills have landed in LDS
    __syncthreads();
  }

  // ---------------- epilogue ----------------
#pragma unroll
  for (int i = 0; i < 2; ++i) {
#pragma unroll
    for (int j = 0; j < 4; ++j) {
      const int rowBase = m0 + waveM * 32 + i * 16 + (lane >> 4) * 8;
      const int col     = n0 + waveN * 64 + j * 16 + (lane & 15);
#pragma unroll
      for (int r = 0; r < 8; ++r) {
        const int row = rowBase + r;
        float v = acc[i][j][r];
        if constexpr (EPI == EPI_PHI || EPI == EPI_PHI_MASK)
          v = (v > 0.0f) ? (v + 1.0f) : __expf(v);
        if constexpr (EPI == EPI_PHI_MASK || EPI == EPI_MASK)
          v = (mask[row] != 0) ? v : 0.0f;
        if constexpr (!TRANS) {
          if constexpr (sizeof(OUTT) == 2)
            reinterpret_cast<unsigned short*>(C)[(size_t)row * ldc + col] = f2bf(v);
          else
            reinterpret_cast<float*>(C)[(size_t)row * ldc + col] = v;
        } else {
          // transposed store: row = b*S + s -> Ct[(b*512 + col)*S + s]  (S=4096)
          const int bb = row >> 12, s = row & 4095;
          reinterpret_cast<unsigned short*>(C)[((size_t)bb * CDKV + col) * CS + s] = f2bf(v);
        }
      }
    }
  }
}

// =====================================================================
// Context kernel: per (b,hkv)  kv[64][64] = K_phi^T V,  z[64] = sum K_phi
//   kT / vT are [(b*512 + d/e), s] bf16 row-major -> fragment-layout-perfect
//   One block per (b,hkv); deterministic (no atomics).
// =====================================================================
__global__ __launch_bounds__(256) void ctx_kernel(
    const unsigned short* __restrict__ kT, const unsigned short* __restrict__ vT,
    float* __restrict__ kvctx, float* __restrict__ zctx) {
  const int bh  = blockIdx.x;            // b*8 + hkv, 32 blocks
  const int tid = threadIdx.x, lane = tid & 31, wid = tid >> 5;
  const int waveM = wid >> 1;            // 4 -> 16 d-rows each
  const int waveNP = wid & 1;            // 2 -> 2 n-subtiles each

  const unsigned short* kb = kT + (size_t)bh * CDH * CS;
  const unsigned short* vb = vT + (size_t)bh * CDH * CS;

  f32x8 acc[2] = {};
  for (int ks = 0; ks < CS / 32; ++ks) {
    const int k0 = ks * 32;
    bf16x16 af;
    {
      const unsigned short* p =
          kb + (size_t)(waveM * 16 + (lane & 15)) * CS + k0 + (lane >> 4) * 8;
      bf16x8 lo = *reinterpret_cast<const bf16x8*>(p);
      bf16x8 hi = *reinterpret_cast<const bf16x8*>(p + 16);
#pragma unroll
      for (int j = 0; j < 8; ++j) { af[j] = lo[j]; af[j + 8] = hi[j]; }
    }
#pragma unroll
    for (int jj = 0; jj < 2; ++jj) {
      const unsigned short* p =
          vb + (size_t)((waveNP * 2 + jj) * 16 + (lane & 15)) * CS + k0 + (lane >> 4) * 8;
      bf16x8 lo = *reinterpret_cast<const bf16x8*>(p);
      bf16x8 hi = *reinterpret_cast<const bf16x8*>(p + 16);
      bf16x16 bfr;
#pragma unroll
      for (int j = 0; j < 8; ++j) { bfr[j] = lo[j]; bfr[j + 8] = hi[j]; }
      acc[jj] = __builtin_amdgcn_wmma_f32_16x16x32_bf16(
          false, af, false, bfr, (short)0, acc[jj], false, false);
    }
  }
  float* kvd = kvctx + (size_t)bh * CDH * CDH;
#pragma unroll
  for (int jj = 0; jj < 2; ++jj) {
#pragma unroll
    for (int r = 0; r < 8; ++r) {
      const int d = waveM * 16 + r + (lane >> 4) * 8;
      const int e = (waveNP * 2 + jj) * 16 + (lane & 15);
      kvd[d * CDH + e] = acc[jj][r];
    }
  }

  // z reduction: 4 threads per d-row, deterministic tree finish
  __shared__ float zred[256];
  {
    const int row = tid >> 2, part = tid & 3;
    const unsigned short* p = kb + (size_t)row * CS;
    float s = 0.f;
    for (int c = part * 32; c < CS; c += 128) {
#pragma unroll
      for (int j = 0; j < 32; j += 8) {
        u16x8 v = *reinterpret_cast<const u16x8*>(p + c + j);
#pragma unroll
        for (int q = 0; q < 8; ++q) s += bf2f(v[q]);
      }
    }
    zred[tid] = s;
  }
  __syncthreads();
  if ((tid & 3) == 0) {
    const int row = tid >> 2;
    zctx[(size_t)bh * CDH + row] =
        zred[tid] + zred[tid + 1] + zred[tid + 2] + zred[tid + 3];
  }
}

// =====================================================================
// Apply kernel: out = (q_phi @ kv) / (q_phi . z + eps), bf16 -> attn
//   grid (B*H, T/128), 256 threads; kv staged transposed bf16 in LDS
// =====================================================================
__global__ __launch_bounds__(256) void apply_kernel(
    const unsigned short* __restrict__ qphi, const float* __restrict__ kvctx,
    const float* __restrict__ zctx, unsigned short* __restrict__ attn) {
  const int bh = blockIdx.x;                   // b*32 + h
  const int b = bh >> 5, h = bh & 31, hkv = h >> 2;
  const int t0 = blockIdx.y * 128;
  const int tid = threadIdx.x, lane = tid & 31, wid = tid >> 5;

  __shared__ __align__(16) unsigned short kvT[64 * 72];  // [e][d], pad 8
  __shared__ float zs[64];
  __shared__ float denL[128];

  const float* kvs = kvctx + (size_t)(b * CHKV + hkv) * CDH * CDH;
  for (int i = tid; i < CDH * CDH; i += 256) {
    const int d = i >> 6, e = i & 63;
    kvT[e * 72 + d] = f2bf(kvs[i]);            // kvs[d*64+e] -> [e][d]
  }
  if (tid < 64) zs[tid] = zctx[(size_t)(b * CHKV + hkv) * CDH + tid];
  __syncthreads();

  // denominator: 2 lanes per row, 32 elems each, shuffle-combine
  {
    const int rowm = lane >> 1, half = lane & 1;
    const size_t qoff =
        ((size_t)(b * CT + t0 + wid * 16 + rowm)) * CD + h * CDH + half * 32;
    float s = 0.f;
#pragma unroll
    for (int j = 0; j < 32; j += 8) {
      u16x8 v = *reinterpret_cast<const u16x8*>(qphi + qoff + j);
#pragma unroll
      for (int q = 0; q < 8; ++q) s += bf2f(v[q]) * zs[half * 32 + j + q];
    }
    s += __shfl_xor(s, 1, 32);
    if (half == 0) denL[wid * 16 + rowm] = s;
  }
  __syncthreads();

  // numerator via WMMA: A = q_phi rows (direct from global), B = kv^T (LDS)
  f32x8 acc[4] = {};
  bf16x16 af[2];
  {
    const size_t qr = ((size_t)(b * CT + t0 + wid * 16 + (lane & 15))) * CD + h * CDH;
#pragma unroll
    for (int kk = 0; kk < 2; ++kk) {
      const unsigned short* p = qphi + qr + kk * 32 + (lane >> 4) * 8;
      bf16x8 lo = *reinterpret_cast<const bf16x8*>(p);
      bf16x8 hi = *reinterpret_cast<const bf16x8*>(p + 16);
#pragma unroll
      for (int j = 0; j < 8; ++j) { af[kk][j] = lo[j]; af[kk][j + 8] = hi[j]; }
    }
  }
#pragma unroll
  for (int jn = 0; jn < 4; ++jn) {
#pragma unroll
    for (int kk = 0; kk < 2; ++kk) {
      const unsigned short* p =
          kvT + (jn * 16 + (lane & 15)) * 72 + kk * 32 + (lane >> 4) * 8;
      bf16x8 lo = *reinterpret_cast<const bf16x8*>(p);
      bf16x8 hi = *reinterpret_cast<const bf16x8*>(p + 16);
      bf16x16 bfr;
#pragma unroll
      for (int j = 0; j < 8; ++j) { bfr[j] = lo[j]; bfr[j + 8] = hi[j]; }
      acc[jn] = __builtin_amdgcn_wmma_f32_16x16x32_bf16(
          false, af[kk], false, bfr, (short)0, acc[jn], false, false);
    }
  }
  // fused divide + bf16 store in final-GEMM-friendly layout (B*T, D)
#pragma unroll
  for (int jn = 0; jn < 4; ++jn) {
    const int col = h * CDH + jn * 16 + (lane & 15);
#pragma unroll
    for (int r = 0; r < 8; ++r) {
      const int mr = wid * 16 + r + (lane >> 4) * 8;
      const float den = denL[mr] + CEPS;
      attn[((size_t)(b * CT + t0 + mr)) * CD + col] = f2bf(acc[jn][r] / den);
    }
  }
}

// =====================================================================
// Launch
// =====================================================================
extern "C" void kernel_launch(void* const* d_in, const int* in_sizes, int n_in,
                              void* d_out, int out_size, void* d_ws, size_t ws_size,
                              hipStream_t stream) {
  const float* x   = (const float*)d_in[0];
  const float* enc = (const float*)d_in[1];
  const int*   msk = (const int*)d_in[2];
  const float* qw  = (const float*)d_in[3];
  const float* kw  = (const float*)d_in[4];
  const float* vw  = (const float*)d_in[5];
  const float* ow  = (const float*)d_in[6];
  float* out = (float*)d_out;

  char* ws = (char*)d_ws;
  size_t off = 0;
  auto alloc = [&](size_t bytes) -> char* {
    char* p = ws + off;
    off = (off + bytes + 255) & ~(size_t)255;
    return p;
  };
  const size_t BT = (size_t)CB * CT;  // 16384 rows

  unsigned short* xb   = (unsigned short*)alloc(BT * CD * 2);
  unsigned short* eb   = (unsigned short*)alloc(BT * CD * 2);
  unsigned short* qwT  = (unsigned short*)alloc((size_t)CD * CD * 2);
  unsigned short* kwT  = (unsigned short*)alloc((size_t)CD * CDKV * 2);
  unsigned short* vwT  = (unsigned short*)alloc((size_t)CD * CDKV * 2);
  unsigned short* owT  = (unsigned short*)alloc((size_t)CD * CD * 2);
  unsigned short* qphi = (unsigned short*)alloc(BT * CD * 2);
  unsigned short* kT   = (unsigned short*)alloc((size_t)CB * CDKV * CS * 2);
  unsigned short* vT   = (unsigned short*)alloc((size_t)CB * CDKV * CS * 2);
  float* kvctx         = (float*)alloc((size_t)CB * CHKV * CDH * CDH * 4);
  float* zctx          = (float*)alloc((size_t)CB * CHKV * CDH * 4);
  unsigned short* attn = (unsigned short*)alloc(BT * CD * 2);

  // 1) convert activations + weights (weights transposed to [N,K])
  {
    size_t n4 = BT * CD / 4;
    cvt_bf16_v4<<<(unsigned)((n4 + 255) / 256), 256, 0, stream>>>(x, xb, n4);
    cvt_bf16_v4<<<(unsigned)((n4 + 255) / 256), 256, 0, stream>>>(enc, eb, n4);
    int knD = CD * CD, knKV = CD * CDKV;
    cvt_transpose_bf16<<<(knD + 255) / 256, 256, 0, stream>>>(qw, qwT, CD, CD);
    cvt_transpose_bf16<<<(knD + 255) / 256, 256, 0, stream>>>(ow, owT, CD, CD);
    cvt_transpose_bf16<<<(knKV + 255) / 256, 256, 0, stream>>>(kw, kwT, CD, CDKV);
    cvt_transpose_bf16<<<(knKV + 255) / 256, 256, 0, stream>>>(vw, vwT, CD, CDKV);
  }

  // 2) q_phi = phi(x @ q_w)              [16384 x 2048]
  wmma_gemm<EPI_PHI, false, unsigned short>
      <<<dim3(CD / BN, (unsigned)(BT / BM)), 256, 0, stream>>>(
          xb, qwT, qphi, nullptr, (int)BT, CD, CD, CD);

  // 3) k_phi^T = (phi(enc @ k_w) * m)^T  and  v^T = (enc @ v_w * m)^T
  wmma_gemm<EPI_PHI_MASK, true, unsigned short>
      <<<dim3(CDKV / BN, (unsigned)(BT / BM)), 256, 0, stream>>>(
          eb, kwT, kT, msk, (int)BT, CDKV, CD, 0);
  wmma_gemm<EPI_MASK, true, unsigned short>
      <<<dim3(CDKV / BN, (unsigned)(BT / BM)), 256, 0, stream>>>(
          eb, vwT, vT, msk, (int)BT, CDKV, CD, 0);

  // 4) contexts kv = K^T V, z = sum K   (one block per (b,hkv))
  ctx_kernel<<<CB * CHKV, 256, 0, stream>>>(kT, vT, kvctx, zctx);

  // 5) attention apply (num/den fused)
  apply_kernel<<<dim3(CB * CH, CT / 128), 256, 0, stream>>>(qphi, kvctx, zctx, attn);

  // 6) out = attn @ o_w  (f32 store)
  wmma_gemm<EPI_NONE, false, float>
      <<<dim3(CD / BN, (unsigned)(BT / BM)), 256, 0, stream>>>(
          attn, owT, out, nullptr, (int)BT, CD, CD, CD);

  (void)in_sizes; (void)n_in; (void)out_size; (void)ws_size;
}
